// FMoE_2834678415367
// MI455X (gfx1250) — matrix operand order repaired and checked by reading, not compiled, not run on hardware
//
#include <hip/hip_runtime.h>
#include <hip/hip_bf16.h>
#include <math.h>

#define N_TOKENS 8192
#define D_MODEL  1024
#define N_EXPERT 8
#define D_HIDDEN 4096
#define TOTAL_PAIRS (N_TOKENS * 2)

#define TM 128          // token rows per block
#define TH 256          // hidden cols per gemm1 block
#define TN 256          // dout cols per gemm2 block
#define KC 64           // K chunk staged in LDS
#define AS_STRIDE 72    // 64 + 8 pad (bf16 elems) -> 144B rows, 16B aligned
#define BS_STRIDE 72
#define CS_STRIDE 264   // 256 + 8 pad
#define ASZ (TM * AS_STRIDE)          // elems per A buffer
#define BSZ (256 * BS_STRIDE)         // elems per B buffer
#define BUFSZ (ASZ + BSZ)             // elems per double-buffer stage

typedef unsigned short ushort_t;
typedef __attribute__((ext_vector_type(16))) __bf16        v16bf;
typedef __attribute__((ext_vector_type(8)))  float         v8f;
typedef __attribute__((ext_vector_type(4)))  unsigned int  v4u;

#if defined(__has_builtin)
#if __has_builtin(__builtin_amdgcn_tensor_load_to_lds) && \
    __has_builtin(__builtin_amdgcn_s_wait_tensorcnt) && \
    __has_builtin(__builtin_amdgcn_groupstaticsize)
#define HAVE_TDM 1
#endif
#endif

#ifdef HAVE_TDM
typedef __attribute__((ext_vector_type(4))) unsigned int tdm_u4;
typedef __attribute__((ext_vector_type(8))) int          tdm_i8;
typedef __attribute__((ext_vector_type(4))) int          tdm_i4;

// Issue one TDM descriptor: load a 2D tile (tile_d0 elems x tile_d1 rows,
// 2-byte elements, row stride stride_elems) from global into LDS, inserting
// 4 DWORDs of pad every 32 DWORDs (= our 64+8 element padded LDS rows).
// D# packing per CDNA5 ISA ch.8 (group0: count/lds_addr/global_addr/type,
// group1: data_size/pad/tensor dims/tile dims/stride). Groups 2/3 unused (2D).
// This toolchain's builtin takes 6 args (g0,g1,g2,g3,g4,cpol) — clang-23 form.
__device__ __forceinline__ void tdm_load_2d(unsigned lds_byte_off, const void* gaddr,
                                            unsigned tile_d0, unsigned tile_d1,
                                            unsigned stride_elems) {
  unsigned long long ga = (unsigned long long)gaddr;
  tdm_u4 g0;
  g0[0] = 1u;                                   // count=1, user mode, no gather
  g0[1] = lds_byte_off;                         // lds_addr (bytes)
  g0[2] = (unsigned)ga;                         // global_addr[31:0]
  g0[3] = (unsigned)(ga >> 32) | (2u << 30);    // global_addr[56:32] | type=2
  tdm_i8 g1;
  g1[0] = (int)((1u << 16)      // data_size = 2 bytes
              | (1u << 20)      // pad_enable
              | (4u << 22)      // pad_interval: 32 DWORDs
              | (3u << 25));    // pad_amount: 4 DWORDs
  g1[1] = (int)((stride_elems & 0xFFFFu) << 16);   // tensor_dim0[15:0]
  g1[2] = 0;                                        // tensor_dim0 hi / tensor_dim1 lo
  g1[3] = (int)(0x10u | (tile_d0 << 16));           // tensor_dim1 hi (1M rows), tile_dim0
  g1[4] = (int)tile_d1;                             // tile_dim1 (tile_dim2 = 0)
  g1[5] = (int)stride_elems;                        // tensor_dim0_stride[31:0]
  g1[6] = 0;
  g1[7] = 0;
  tdm_i4 z4 = {0, 0, 0, 0};
  tdm_i8 z8 = {0, 0, 0, 0, 0, 0, 0, 0};
  __builtin_amdgcn_tensor_load_to_lds(g0, g1, z4, z4, z8, 0);
}
#endif

__device__ __forceinline__ ushort_t f2bf(float f) {
  union { float f; unsigned int u; } v; v.f = f;
  unsigned int r = v.u + 0x7FFFu + ((v.u >> 16) & 1u);   // RNE
  return (ushort_t)(r >> 16);
}

// Branch-free tanh-gelu: prefer gfx1250 native V_TANH_F32.
__device__ __forceinline__ float gelu_f(float x) {
  float z = 0.7978845608028654f * (x + 0.044715f * x * x * x);
#if __has_builtin(__builtin_amdgcn_tanhf)
  return 0.5f * x * (1.0f + __builtin_amdgcn_tanhf(z));
#else
  return x * __builtin_amdgcn_rcpf(1.0f + __expf(-2.0f * z));
#endif
}

// Build a 16-element bf16 WMMA fragment from two 16B LDS loads.
__device__ __forceinline__ v16bf frag_ld(const ushort_t* p0, const ushort_t* p1) {
  union { v16bf v; v4u q[2]; } f;
  f.q[0] = *(const v4u*)p0;
  f.q[1] = *(const v4u*)p1;
  return f.v;
}

// ---------------- init ----------------
__global__ void zero_kernel(float* y, long n4, int* cnt) {
  long i = (long)blockIdx.x * blockDim.x + threadIdx.x;
  long stride = (long)gridDim.x * blockDim.x;
  float4 z = make_float4(0.f, 0.f, 0.f, 0.f);
  for (long k = i; k < n4; k += stride) ((float4*)y)[k] = z;
  if (blockIdx.x == 0 && threadIdx.x < N_EXPERT) cnt[threadIdx.x] = 0;
}

__global__ void cvt_bf16_kernel(const float* __restrict__ in, ushort_t* __restrict__ out, long n) {
  long i = ((long)blockIdx.x * blockDim.x + threadIdx.x) * 4;
  if (i >= n) return;
  float4 v = *(const float4*)(in + i);
  ushort4 o;
  o.x = f2bf(v.x); o.y = f2bf(v.y); o.z = f2bf(v.z); o.w = f2bf(v.w);
  *(ushort4*)(out + i) = o;
}

// in: [R][C] fp32  ->  out: [C][R] bf16   (per-matrix blockIdx.z)
__global__ void cvt_transpose_kernel(const float* __restrict__ in, ushort_t* __restrict__ out,
                                     int R, int C) {
  __shared__ float tile[32][33];
  const float* src = in  + (size_t)blockIdx.z * R * C;
  ushort_t*    dst = out + (size_t)blockIdx.z * R * C;
  int c0 = blockIdx.x * 32, r0 = blockIdx.y * 32;
  #pragma unroll
  for (int k = 0; k < 32; k += 8)
    tile[threadIdx.y + k][threadIdx.x] = src[(size_t)(r0 + threadIdx.y + k) * C + c0 + threadIdx.x];
  __syncthreads();
  #pragma unroll
  for (int k = 0; k < 32; k += 8)
    dst[(size_t)(c0 + threadIdx.y + k) * R + r0 + threadIdx.x] = f2bf(tile[threadIdx.x][threadIdx.y + k]);
}

// ---------------- gate: logits -> top2 -> softmax -> bucket append ----------------
__global__ void gate_kernel(const float* __restrict__ inp, const float* __restrict__ gw,
                            const float* __restrict__ gb,
                            int* cnt, int* tok_list, float* tok_score) {
  int wave = threadIdx.x >> 5;
  int lane = threadIdx.x & 31;
  int t = blockIdx.x * 8 + wave;
  float acc[N_EXPERT];
  #pragma unroll
  for (int e = 0; e < N_EXPERT; e++) acc[e] = 0.f;
  const float* x = inp + (size_t)t * D_MODEL;
  for (int d = lane; d < D_MODEL; d += 32) {
    float xv = x[d];
    float4 g0 = *(const float4*)(gw + d * N_EXPERT);
    float4 g1 = *(const float4*)(gw + d * N_EXPERT + 4);
    acc[0] += xv * g0.x; acc[1] += xv * g0.y; acc[2] += xv * g0.z; acc[3] += xv * g0.w;
    acc[4] += xv * g1.x; acc[5] += xv * g1.y; acc[6] += xv * g1.z; acc[7] += xv * g1.w;
  }
  #pragma unroll
  for (int e = 0; e < N_EXPERT; e++) {
    #pragma unroll
    for (int m = 16; m >= 1; m >>= 1) acc[e] += __shfl_xor(acc[e], m, 32);
  }
  if (lane == 0) {
    #pragma unroll
    for (int e = 0; e < N_EXPERT; e++) acc[e] += gb[e];
    int e0 = 0; float v0 = acc[0];
    for (int e = 1; e < N_EXPERT; e++) if (acc[e] > v0) { v0 = acc[e]; e0 = e; }
    int e1 = (e0 == 0) ? 1 : 0; float v1 = acc[e1];
    for (int e = 0; e < N_EXPERT; e++) if (e != e0 && acc[e] > v1) { v1 = acc[e]; e1 = e; }
    float p0 = 1.0f / (1.0f + __expf(v1 - v0));
    float p1 = 1.0f - p0;
    int s0 = atomicAdd(&cnt[e0], 1);
    tok_list[e0 * N_TOKENS + s0] = t; tok_score[e0 * N_TOKENS + s0] = p0;
    int s1 = atomicAdd(&cnt[e1], 1);
    tok_list[e1 * N_TOKENS + s1] = t; tok_score[e1 * N_TOKENS + s1] = p1;
  }
}

__global__ void prefix_kernel(const int* cnt, int* off) {
  if (threadIdx.x == 0) {
    int a = 0;
    for (int e = 0; e < N_EXPERT; e++) { off[e] = a; a += cnt[e]; }
  }
}

// ---------------- gemm1: H = gelu(X @ W1 + b1), bf16 out ----------------
// 512 threads / 16 waves; C tile 128x256; each wave: 2 M-tiles x 4 N-tiles.
// Double-buffered LDS; W1 tile staged by the Tensor Data Mover (TENSORcnt),
// gathered X rows staged by the waves. TDM issue is gated by a *scalar*
// branch (readfirstlane) because TDM ignores EXEC — exec-predication alone
// would make every wave issue a duplicate DMA.
__global__ __launch_bounds__(512) void gemm1_kernel(
    const ushort_t* __restrict__ xb, const ushort_t* __restrict__ w1t,
    const float* __restrict__ b1,
    const int* __restrict__ cnt, const int* __restrict__ off,
    const int* __restrict__ tok_list, ushort_t* __restrict__ Hbuf) {
  const int e = blockIdx.z;
  const int ce = cnt[e];
  const int row0 = blockIdx.x * TM;
  if (row0 >= ce) return;
  const int base = off[e];
  const int h0 = blockIdx.y * TH;

  extern __shared__ char smem[];
  ushort_t* lds = (ushort_t*)smem;                    // 2 x (As + Bs)
  int* tl = (int*)(lds + 2 * BUFSZ);                  // 128 tokens

  const int tid = threadIdx.x;
  const int w = tid >> 5, lane = tid & 31;
  const int mg = w & 3, ng = w >> 2;
  const int lhalf = lane >> 4, l15 = lane & 15;

  if (tid < TM) tl[tid] = (row0 + tid < ce) ? tok_list[e * N_TOKENS + row0 + tid] : 0;
  __syncthreads();

  const ushort_t* w1e = w1t + (size_t)e * D_HIDDEN * D_MODEL;
  v8f acc[2][4] = {};

  auto stageA = [&](int kk, int b) {            // X tile 128x64 (gathered rows)
    ushort_t* As = lds + b * BUFSZ;
    #pragma unroll
    for (int k2 = 0; k2 < 2; k2++) {
      int idx = tid + k2 * 512;
      int r = idx >> 3, seg = idx & 7;
      *(v4u*)(As + r * AS_STRIDE + seg * 8) =
          *(const v4u*)(xb + (size_t)tl[r] * D_MODEL + kk + seg * 8);
    }
  };
#ifdef HAVE_TDM
  const unsigned ldsdyn = __builtin_amdgcn_groupstaticsize();
  // scalar (SGPR) wave-0 test -> s_cmp + s_cbranch, not exec predication
  const bool tdm_issuer = (__builtin_amdgcn_readfirstlane(tid) == 0);
  auto stageB = [&](int kk, int b) {            // W1^T tile 256x64 via TDM
    if (tdm_issuer)
      tdm_load_2d(ldsdyn + (unsigned)((b * BUFSZ + ASZ) * 2),
                  w1e + (size_t)h0 * D_MODEL + kk, KC, TH, D_MODEL);
  };
#else
  auto stageB = [&](int kk, int b) {
    ushort_t* Bs = lds + b * BUFSZ + ASZ;
    int hr = tid >> 1, half = tid & 1;
    const ushort_t* src = w1e + (size_t)(h0 + hr) * D_MODEL + kk + half * 32;
    ushort_t* dst = Bs + hr * BS_STRIDE + half * 32;
    #pragma unroll
    for (int q = 0; q < 4; q++) *(v4u*)(dst + q * 8) = *(const v4u*)(src + q * 8);
  };
#endif

  stageA(0, 0);
  stageB(0, 0);
  const int NCHUNK = D_MODEL / KC;              // 16
  for (int c = 0; c < NCHUNK; c++) {
#ifdef HAVE_TDM
    // scalar wait, unconditional: waves that issued nothing have TENSORcnt==0
    __builtin_amdgcn_s_wait_tensorcnt(0);
#endif
    __syncthreads();
    if (c + 1 < NCHUNK) { stageA((c + 1) * KC, (c + 1) & 1); stageB((c + 1) * KC, (c + 1) & 1); }
    const ushort_t* As = lds + (c & 1) * BUFSZ;
    const ushort_t* Bs = As + ASZ;
    #pragma unroll
    for (int ks = 0; ks < KC; ks += 32) {
      v16bf a[2];
      #pragma unroll
      for (int i = 0; i < 2; i++) {
        const ushort_t* ap = As + ((mg * 2 + i) * 16 + l15) * AS_STRIDE + ks + lhalf * 8;
        a[i] = frag_ld(ap, ap + 16);            // K 0..7 / 16..23 (hi lanes +8)
      }
      #pragma unroll
      for (int j = 0; j < 4; j++) {
        const ushort_t* bp = Bs + ((ng * 4 + j) * 16 + l15) * BS_STRIDE + ks + lhalf * 8;
        v16bf b = frag_ld(bp, bp + 16);         // K 0..7 / 16..23 (hi lanes +8)
        #pragma unroll
        for (int i = 0; i < 2; i++)
          acc[i][j] = __builtin_amdgcn_wmma_f32_16x16x32_bf16(
              false, a[i], false, b, (short)0, acc[i][j], false, false);
      }
    }
  }
  __syncthreads();                              // done reading LDS buffers

  // epilogue: bias + gelu -> bf16 tile in LDS -> coalesced global H store
  ushort_t* Cs = lds;                           // 128 x 264, fits in buffers
  #pragma unroll
  for (int i = 0; i < 2; i++) {
    int mrow = (mg * 2 + i) * 16;
    #pragma unroll
    for (int j = 0; j < 4; j++) {
      int col = (ng * 4 + j) * 16 + l15;
      float bias = b1[(size_t)e * D_HIDDEN + h0 + col];
      #pragma unroll
      for (int r = 0; r < 8; r++) {
        int m = mrow + r + lhalf * 8;           // C layout: VGPR r -> M=r (+8 hi lanes)
        Cs[m * CS_STRIDE + col] = f2bf(gelu_f(acc[i][j][r] + bias));
      }
    }
  }
  __syncthreads();
  #pragma unroll
  for (int k2 = 0; k2 < 2; k2++) {
    int idx = tid + k2 * 512;
    int r = idx >> 3, seg = idx & 7;
    // ragged tail rows go to a trash region past TOTAL_PAIRS (never read)
    size_t grow = (row0 + r < ce) ? (size_t)(base + row0 + r) : (size_t)(TOTAL_PAIRS + r);
    ushort_t* dst = Hbuf + grow * D_HIDDEN + h0 + seg * 32;
    const ushort_t* src = Cs + r * CS_STRIDE + seg * 32;
    #pragma unroll
    for (int q = 0; q < 4; q++) *(v4u*)(dst + q * 8) = *(const v4u*)(src + q * 8);
  }
}

// ---------------- gemm2: y += score * (H @ W2 + b2) ----------------
// Both tiles are dense 2D -> both staged by TDM; waves do pure ds_load+WMMA.
__global__ __launch_bounds__(512) void gemm2_kernel(
    const ushort_t* __restrict__ Hbuf, const ushort_t* __restrict__ w2t,
    const float* __restrict__ b2,
    const int* __restrict__ cnt, const int* __restrict__ off,
    const int* __restrict__ tok_list, const float* __restrict__ tok_score,
    float* __restrict__ y) {
  const int e = blockIdx.z;
  const int ce = cnt[e];
  const int row0 = blockIdx.x * TM;
  if (row0 >= ce) return;
  const int base = off[e];
  const int n0 = blockIdx.y * TN;

  extern __shared__ char smem[];
  ushort_t* lds = (ushort_t*)smem;
  int*   tl = (int*)(lds + 2 * BUFSZ);
  float* ts = (float*)(tl + TM);

  const int tid = threadIdx.x;
  const int w = tid >> 5, lane = tid & 31;
  const int mg = w & 3, ng = w >> 2;
  const int lhalf = lane >> 4, l15 = lane & 15;

  if (tid < TM) {
    bool valid = (row0 + tid) < ce;
    tl[tid] = valid ? tok_list[e * N_TOKENS + row0 + tid] : 0;
    ts[tid] = valid ? tok_score[e * N_TOKENS + row0 + tid] : 0.0f;
  }
  __syncthreads();

  const ushort_t* w2e = w2t + (size_t)e * D_MODEL * D_HIDDEN;
  const ushort_t* hrow = Hbuf + (size_t)(base + row0) * D_HIDDEN;  // dense slot rows
  v8f acc[2][4] = {};

#ifdef HAVE_TDM
  const unsigned ldsdyn = __builtin_amdgcn_groupstaticsize();
  const bool tdm_issuer = (__builtin_amdgcn_readfirstlane(tid) == 0);
  auto stage = [&](int kk, int b) {
    if (tdm_issuer) {
      tdm_load_2d(ldsdyn + (unsigned)((b * BUFSZ) * 2),
                  hrow + kk, KC, TM, D_HIDDEN);                 // H tile 128x64
      tdm_load_2d(ldsdyn + (unsigned)((b * BUFSZ + ASZ) * 2),
                  w2e + (size_t)n0 * D_HIDDEN + kk, KC, TN, D_HIDDEN);  // W2^T 256x64
    }
  };
#else
  auto stage = [&](int kk, int b) {
    ushort_t* As = lds + b * BUFSZ;
    ushort_t* Bs = As + ASZ;
    #pragma unroll
    for (int k2 = 0; k2 < 2; k2++) {
      int idx = tid + k2 * 512;
      int r = idx >> 3, seg = idx & 7;
      size_t grow = base + ((row0 + r < ce) ? (row0 + r) : row0);
      *(v4u*)(As + r * AS_STRIDE + seg * 8) =
          *(const v4u*)(Hbuf + grow * D_HIDDEN + kk + seg * 8);
    }
    int nr = tid >> 1, half = tid & 1;
    const ushort_t* src = w2e + (size_t)(n0 + nr) * D_HIDDEN + kk + half * 32;
    ushort_t* dst = Bs + nr * BS_STRIDE + half * 32;
    #pragma unroll
    for (int q = 0; q < 4; q++) *(v4u*)(dst + q * 8) = *(const v4u*)(src + q * 8);
  };
#endif

  stage(0, 0);
  const int NCHUNK = D_HIDDEN / KC;             // 64
  for (int c = 0; c < NCHUNK; c++) {
#ifdef HAVE_TDM
    __builtin_amdgcn_s_wait_tensorcnt(0);
#endif
    __syncthreads();
    if (c + 1 < NCHUNK) stage((c + 1) * KC, (c + 1) & 1);
    const ushort_t* As = lds + (c & 1) * BUFSZ;
    const ushort_t* Bs = As + ASZ;
    #pragma unroll
    for (int ks = 0; ks < KC; ks += 32) {
      v16bf a[2];
      #pragma unroll
      for (int i = 0; i < 2; i++) {
        const ushort_t* ap = As + ((mg * 2 + i) * 16 + l15) * AS_STRIDE + ks + lhalf * 8;
        a[i] = frag_ld(ap, ap + 16);
      }
      #pragma unroll
      for (int j = 0; j < 4; j++) {
        const ushort_t* bp = Bs + ((ng * 4 + j) * 16 + l15) * BS_STRIDE + ks + lhalf * 8;
        v16bf b = frag_ld(bp, bp + 16);
        #pragma unroll
        for (int i = 0; i < 2; i++)
          acc[i][j] = __builtin_amdgcn_wmma_f32_16x16x32_bf16(
              false, a[i], false, b, (short)0, acc[i][j], false, false);
      }
    }
  }

  #pragma unroll
  for (int i = 0; i < 2; i++) {
    int mrow = (mg * 2 + i) * 16;
    float sv[8]; int tv[8];
    #pragma unroll
    for (int r = 0; r < 8; r++) {
      int m = mrow + r + lhalf * 8;
      sv[r] = ts[m];
      tv[r] = tl[m];
    }
    #pragma unroll
    for (int j = 0; j < 4; j++) {
      int col = n0 + (ng * 4 + j) * 16 + l15;
      float bias = b2[(size_t)e * D_MODEL + col];
      #pragma unroll
      for (int r = 0; r < 8; r++)
        unsafeAtomicAdd(y + (size_t)tv[r] * D_MODEL + col, sv[r] * (acc[i][j][r] + bias));
    }
  }
}

// ---------------- launch ----------------
extern "C" void kernel_launch(void* const* d_in, const int* in_sizes, int n_in,
                              void* d_out, int out_size, void* d_ws, size_t ws_size,
                              hipStream_t stream) {
  const float* inp    = (const float*)d_in[0];
  const float* gate_w = (const float*)d_in[1];
  const float* gate_b = (const float*)d_in[2];
  const float* w1     = (const float*)d_in[3];
  const float* b1     = (const float*)d_in[4];
  const float* w2     = (const float*)d_in[5];
  const float* b2     = (const float*)d_in[6];
  float* y = (float*)d_out;
  char* ws = (char*)d_ws;

  size_t o = 0;
  ushort_t* xb   = (ushort_t*)(ws + o); o += (size_t)N_TOKENS * D_MODEL * 2;
  ushort_t* w1t  = (ushort_t*)(ws + o); o += (size_t)N_EXPERT * D_HIDDEN * D_MODEL * 2;
  ushort_t* w2t  = (ushort_t*)(ws + o); o += (size_t)N_EXPERT * D_MODEL * D_HIDDEN * 2;
  ushort_t* Hbuf = (ushort_t*)(ws + o); o += (size_t)(TOTAL_PAIRS + TM) * D_HIDDEN * 2;
  int*   cnt       = (int*)(ws + o);   o += 64;
  int*   off       = (int*)(ws + o);   o += 64;
  int*   tok_list  = (int*)(ws + o);   o += (size_t)N_EXPERT * N_TOKENS * 4;
  float* tok_score = (float*)(ws + o); o += (size_t)N_EXPERT * N_TOKENS * 4;

  zero_kernel<<<2048, 256, 0, stream>>>(y, (long)N_TOKENS * D_MODEL / 4, cnt);
  cvt_bf16_kernel<<<(N_TOKENS * D_MODEL / 4 + 255) / 256, 256, 0, stream>>>(
      inp, xb, (long)N_TOKENS * D_MODEL);
  cvt_transpose_kernel<<<dim3(D_HIDDEN / 32, D_MODEL / 32, N_EXPERT), dim3(32, 8), 0, stream>>>(
      w1, w1t, D_MODEL, D_HIDDEN);
  cvt_transpose_kernel<<<dim3(D_MODEL / 32, D_HIDDEN / 32, N_EXPERT), dim3(32, 8), 0, stream>>>(
      w2, w2t, D_HIDDEN, D_MODEL);
  gate_kernel<<<N_TOKENS / 8, 256, 0, stream>>>(inp, gate_w, gate_b, cnt, tok_list, tok_score);
  prefix_kernel<<<1, 32, 0, stream>>>(cnt, off);

  size_t ldsz = (size_t)(2 * BUFSZ) * 2 + TM * 4 + TM * 4;   // ~112 KB
  gemm1_kernel<<<dim3(N_TOKENS / TM, D_HIDDEN / TH, N_EXPERT), 512, ldsz, stream>>>(
      xb, w1t, b1, cnt, off, tok_list, Hbuf);
  gemm2_kernel<<<dim3(N_TOKENS / TM, D_MODEL / TN, N_EXPERT), 512, ldsz, stream>>>(
      Hbuf, w2t, b2, cnt, off, tok_list, tok_score, y);
}
